// MHA_25426206392622
// MI455X (gfx1250) — matrix operand order repaired and checked by reading, not compiled, not run on hardware
//
#include <hip/hip_runtime.h>
#include <hip/hip_bf16.h>

// ---------------------------------------------------------------------------
// MHA forward for gfx1250 (MI455X): bf16 WMMA + async-to-LDS staging.
// Flash attention computes S^T = K*Q^T so that the softmax is per-lane and the
// probability matrix is already in WMMA A-fragment layout (no LDS round-trip).
// ---------------------------------------------------------------------------

constexpr int kB = 2;
constexpr int kS = 2048;
constexpr int kD = 1024;
constexpr int kH = 16;
constexpr int kHD = kD / kH;   // 64

typedef __attribute__((ext_vector_type(8)))  __bf16 v8bf;
typedef __attribute__((ext_vector_type(16))) __bf16 v16bf;
typedef __attribute__((ext_vector_type(8)))  float  v8f;

__device__ __forceinline__ unsigned short f32_to_bf16(float f) {
    unsigned int u = __float_as_uint(f);
    u += 0x7FFFu + ((u >> 16) & 1u);     // round-to-nearest-even
    return (unsigned short)(u >> 16);
}

// Build a v16bf WMMA operand fragment from two 16-byte chunks.
__device__ __forceinline__ v16bf frag16(const unsigned short* p0, const unsigned short* p1) {
    v8bf lo = *(const v8bf*)p0;
    v8bf hi = *(const v8bf*)p1;
    return __builtin_shufflevector(lo, hi, 0,1,2,3,4,5,6,7,8,9,10,11,12,13,14,15);
}

// v_perm_b32 16-bit interleaves: one instruction per packed word.
__device__ __forceinline__ unsigned perm_pack_lo(unsigned a, unsigned b) {
    return __builtin_amdgcn_perm(b, a, 0x05040100u);  // a[15:0] | b[15:0]<<16
}
__device__ __forceinline__ unsigned perm_pack_hi(unsigned a, unsigned b) {
    return __builtin_amdgcn_perm(b, a, 0x07060302u);  // a[31:16] | b[31:16]<<16
}

// ---- async copy: 16 bytes global -> LDS, tracked with ASYNCcnt -------------
__device__ __forceinline__ void async_b128(const void* gptr, unsigned lds_addr) {
    asm volatile("global_load_async_to_lds_b128 %0, %1, off"
                 :: "v"(lds_addr), "v"(gptr) : "memory");
}
__device__ __forceinline__ unsigned lds_addr_of(const void* lptr) {
    return (unsigned)(uintptr_t)lptr;    // addr[31:0] == LDS offset
}
__device__ __forceinline__ void async_wait0() {
    asm volatile("s_wait_asynccnt 0x0" ::: "memory");
}

// ---------------------------------------------------------------------------
// Pre-pass 1: fp32 -> bf16 (x activations), 8 elements/thread
// ---------------------------------------------------------------------------
__global__ __launch_bounds__(256)
void cvt_f32_bf16(const float* __restrict__ in, unsigned short* __restrict__ out, int n) {
    int idx = (blockIdx.x * 256 + threadIdx.x) * 8;
    if (idx >= n) return;
    float4 a = *(const float4*)&in[idx];
    float4 b = *(const float4*)&in[idx + 4];
    union { unsigned short us[8]; uint4 u; } p;
    p.us[0] = f32_to_bf16(a.x); p.us[1] = f32_to_bf16(a.y);
    p.us[2] = f32_to_bf16(a.z); p.us[3] = f32_to_bf16(a.w);
    p.us[4] = f32_to_bf16(b.x); p.us[5] = f32_to_bf16(b.y);
    p.us[6] = f32_to_bf16(b.z); p.us[7] = f32_to_bf16(b.w);
    *(uint4*)&out[idx] = p.u;
}

// ---------------------------------------------------------------------------
// Pre-pass 2: fp32 [K][N] -> bf16 [N][K] (weights, transposed once)
// ---------------------------------------------------------------------------
__global__ __launch_bounds__(256)
void transpose_cvt(const float* __restrict__ in, unsigned short* __restrict__ out,
                   int K, int N) {
    __shared__ float tile[32][33];
    const int tid = threadIdx.x;
    const int n0 = blockIdx.x * 32, k0 = blockIdx.y * 32;
    {
        int e = tid * 4;
        int r = e >> 5, c = e & 31;            // r: k-within, c: n-within
        const float4 v = *(const float4*)&in[(size_t)(k0 + r) * N + n0 + c];
        tile[r][c] = v.x; tile[r][c + 1] = v.y; tile[r][c + 2] = v.z; tile[r][c + 3] = v.w;
    }
    __syncthreads();
    {
        int e = tid * 4;
        int nn = e >> 5, kk = e & 31;
        union { unsigned short us[4]; uint2 u; } p;
#pragma unroll
        for (int j = 0; j < 4; ++j) p.us[j] = f32_to_bf16(tile[kk + j][nn]);
        *(uint2*)&out[(size_t)(n0 + nn) * K + k0 + kk] = p.u;
    }
}

// ---------------------------------------------------------------------------
// Tiled GEMM: C[M,N] = A[M,K] * Bt[N,K]^T, A/Bt bf16 (pre-converted),
// TO = unsigned short (bf16 out) or float.
// 256 threads / 8 waves (4x2), block tile 256x128, wave tile 64x64, K-step 32.
// Async-to-LDS staging; 16 WMMAs per wave per K-step.
// ---------------------------------------------------------------------------
template <typename TO>
__global__ __launch_bounds__(256)
void gemm_wmma_bf16(const unsigned short* __restrict__ A,   // [M][K] bf16
                    const unsigned short* __restrict__ Bt,  // [N][K] bf16
                    TO* __restrict__ C, int M, int N, int K) {
    __shared__ __align__(16) unsigned short As[256 * 32];   // [row][k] 16KB
    __shared__ __align__(16) unsigned short Bs[128 * 32];   // [n][k]    8KB

    const int tid  = threadIdx.x;
    const int wave = tid >> 5;
    const int lane = tid & 31;
    const int wm   = wave >> 1;          // 0..3 -> 64-row slabs
    const int wn   = wave & 1;           // 0..1 -> 64-col slabs
    const int m0   = blockIdx.y * 256;
    const int n0   = blockIdx.x * 128;
    const int row  = lane & 15;
    const int hiK  = lane >> 4;
    const int kb   = hiK ? 8 : 0;        // A-fragment K start
    const int kk   = hiK * 8;            // B-fragment K start

    // ---- loop-invariant staging pointers (bump by 32 elems per K-step) ----
    const unsigned short* gap[4]; unsigned lap[4];
#pragma unroll
    for (int i = 0; i < 4; ++i) {
        int e = (tid + i * 256) * 8;
        int r_ = e >> 5, c_ = e & 31;
        gap[i] = &A[(size_t)(m0 + r_) * K + c_];
        lap[i] = lds_addr_of(&As[r_ * 32 + c_]);
    }
    const unsigned short* gbp[2]; unsigned lbp[2];
#pragma unroll
    for (int i = 0; i < 2; ++i) {
        int e = (tid + i * 256) * 8;
        int r_ = e >> 5, c_ = e & 31;
        gbp[i] = &Bt[(size_t)(n0 + r_) * K + c_];
        lbp[i] = lds_addr_of(&Bs[r_ * 32 + c_]);
    }
    // ---- loop-invariant fragment pointers ----
    const unsigned short* pA[4];
    const unsigned short* pB[4];
#pragma unroll
    for (int t = 0; t < 4; ++t) pA[t] = &As[(wm * 64 + t * 16 + row) * 32 + kb];
#pragma unroll
    for (int u = 0; u < 4; ++u) pB[u] = &Bs[(wn * 64 + u * 16 + row) * 32 + kk];

    v8f acc[4][4];
#pragma unroll
    for (int t = 0; t < 4; ++t)
#pragma unroll
        for (int u = 0; u < 4; ++u) acc[t][u] = (v8f)0.0f;

    for (int k0 = 0; k0 < K; k0 += 32) {
        // ---- async stage A (4x16B) & B (2x16B) tiles ----
#pragma unroll
        for (int i = 0; i < 4; ++i) { async_b128(gap[i], lap[i]); gap[i] += 32; }
#pragma unroll
        for (int i = 0; i < 2; ++i) { async_b128(gbp[i], lbp[i]); gbp[i] += 32; }
        if (k0 + 32 < K) {
            __builtin_prefetch(gap[0], 0, 1);
            __builtin_prefetch(gbp[0], 0, 1);
        }
        async_wait0();
        __syncthreads();

        // ---- fragments + 16 WMMAs ----
        v16bf af[4], bf[4];
#pragma unroll
        for (int t = 0; t < 4; ++t) af[t] = frag16(pA[t], pA[t] + 16);
#pragma unroll
        for (int u = 0; u < 4; ++u) bf[u] = frag16(pB[u], pB[u] + 16);
#pragma unroll
        for (int t = 0; t < 4; ++t)
#pragma unroll
            for (int u = 0; u < 4; ++u)
                acc[t][u] = __builtin_amdgcn_wmma_f32_16x16x32_bf16(
                    false, af[t], false, bf[u], (short)0, acc[t][u], false, false);
        __syncthreads();
    }

    // ---- store: C layout lane = column, rows r (+8 for hi lanes) ----
#pragma unroll
    for (int t = 0; t < 4; ++t) {
#pragma unroll
        for (int u = 0; u < 4; ++u) {
            int gn = n0 + wn * 64 + u * 16 + row;
#pragma unroll
            for (int r = 0; r < 8; ++r) {
                int gm = m0 + wm * 64 + t * 16 + r + hiK * 8;
                if constexpr (sizeof(TO) == 2)
                    C[(size_t)gm * N + gn] = f32_to_bf16(acc[t][u][r]);
                else
                    C[(size_t)gm * N + gn] = acc[t][u][r];
            }
        }
    }
}

// ---------------------------------------------------------------------------
// Flash attention, S^T formulation.
// Block = 128 threads (4 waves) per 64 query rows per (b,h); wave owns 16 rows.
//   S^T tile = K_tile(16x32hd) x Q^T(32hd x 16q):  lane = query, regs = keys
//   -> softmax stats are per-lane scalars (one __shfl_xor(16) per reduction)
//   -> P^T C-layout == A-fragment layout of P for O = P x V (no LDS staging)
// ---------------------------------------------------------------------------
__global__ __launch_bounds__(128)
void flash_attn(const unsigned short* __restrict__ qkv,   // [B,S,3D] bf16
                unsigned short* __restrict__ o) {         // [B,S,D]  bf16
    __shared__ __align__(16) unsigned short kt[32 * 64];  // [key][hd]
    __shared__ __align__(16) unsigned short vt[64 * 32];  // [hd][key]
    __shared__ __align__(16) float bc[4][16];             // per-wave row broadcast

    const int tid   = threadIdx.x;
    const int wave  = tid >> 5;
    const int lane  = tid & 31;
    const int qtile = blockIdx.x;
    const int bh    = blockIdx.y;
    const int b     = bh / kH;
    const int h     = bh % kH;

    const int q_base = qtile * 64 + wave * 16;
    const int col    = lane & 15;        // query col (S^T) / hd col (O)
    const int hiK    = lane >> 4;
    const int base8  = hiK * 8;
    const int kb     = hiK ? 8 : 0;      // A-fragment K start
    const int kk     = hiK * 8;          // B-fragment K start
    const int q_g    = q_base + col;     // this lane's query row

    // Q^T as B operand: lane = query column, hd contiguous; loaded once.
    v16bf qb[2];
    {
        const size_t rbase = ((size_t)b * kS + q_g) * (3 * kD) + h * kHD;
#pragma unroll
        for (int s = 0; s < 2; ++s) {
            const unsigned short* p0 = qkv + rbase + s * 32 + kk;
            qb[s] = frag16(p0, p0 + 16);
        }
    }

    // ---- loop-invariant staging pointers (bump by 32 rows per K-tile) ----
    const ptrdiff_t kstep = (ptrdiff_t)32 * 3 * kD;
    const unsigned short* gkp[2]; unsigned lk[2];
#pragma unroll
    for (int i = 0; i < 2; ++i) {
        int linear = tid + i * 128;
        int key = linear >> 3;
        int d0  = (linear & 7) * 8;
        gkp[i] = &qkv[((size_t)b * kS + key) * (3 * kD) + kD + h * kHD + d0];
        lk[i]  = lds_addr_of(&kt[key * 64 + d0]);
    }
    const int kp  = (tid & 15) * 2;      // key pair for V transpose
    const int d0v = (tid >> 4) * 8;      // 8 hd values
    const unsigned short* gvp =
        &qkv[((size_t)b * kS + kp) * (3 * kD) + 2 * kD + h * kHD + d0v];
    // loop-invariant fragment pointers
    const unsigned short* pK[2][2];
#pragma unroll
    for (int u = 0; u < 2; ++u)
#pragma unroll
        for (int s = 0; s < 2; ++s)
            pK[u][s] = &kt[(u * 16 + col) * 64 + s * 32 + kb];
    const unsigned short* pV[4];
#pragma unroll
    for (int u = 0; u < 4; ++u) pV[u] = &vt[(u * 16 + col) * 32 + kk];

    const float scale = 0.125f;          // 1/sqrt(64)
    float m_run = -1e30f, l_run = 0.0f;  // per-lane (= per-query) stats
    v8f acc[4];
#pragma unroll
    for (int u = 0; u < 4; ++u) acc[u] = (v8f)0.0f;

    const int q_end = qtile * 64 + 63;
    for (int k0 = 0; k0 <= q_end; k0 += 32) {
        // ---- stage K tile [32][64] via async; V tile [64][32] via perm-pack
        async_b128(gkp[0], lk[0]); gkp[0] += kstep;
        async_b128(gkp[1], lk[1]); gkp[1] += kstep;
        {
            uint4 va = *(const uint4*)gvp;
            uint4 vb = *(const uint4*)(gvp + 3 * kD);
            gvp += kstep;
            const unsigned* aw = (const unsigned*)&va;
            const unsigned* bw = (const unsigned*)&vb;
#pragma unroll
            for (int w = 0; w < 4; ++w) {
                *(unsigned*)&vt[(d0v + 2 * w)     * 32 + kp] = perm_pack_lo(aw[w], bw[w]);
                *(unsigned*)&vt[(d0v + 2 * w + 1) * 32 + kp] = perm_pack_hi(aw[w], bw[w]);
            }
        }
        async_wait0();
        __syncthreads();

        // ---- S^T: two 16-key tiles, 2 hd-steps each (4 WMMAs) ----
        v8f st[2];
        st[0] = (v8f)0.0f; st[1] = (v8f)0.0f;
#pragma unroll
        for (int u = 0; u < 2; ++u)
#pragma unroll
            for (int s = 0; s < 2; ++s) {
                v16bf ak = frag16(pK[u][s], pK[u][s] + 16);
                st[u] = __builtin_amdgcn_wmma_f32_16x16x32_bf16(
                    false, ak, false, qb[s], (short)0, st[u], false, false);
            }

        // ---- per-lane softmax over 32 keys (regs) + one cross-half shuffle --
        float p0v[8], p1v[8];
        float mt = -1e30f;
#pragma unroll
        for (int r = 0; r < 8; ++r) {
            float s0 = st[0][r] * scale;
            float s1 = st[1][r] * scale;
            if (k0 + base8 + r > q_g)      s0 = -1e30f;   // causal mask
            if (k0 + 16 + base8 + r > q_g) s1 = -1e30f;
            p0v[r] = s0; p1v[r] = s1;
            mt = fmaxf(mt, fmaxf(s0, s1));
        }
        mt = fmaxf(mt, __shfl_xor(mt, 16, 32));
        float mn  = fmaxf(m_run, mt);
        float fac = __expf(m_run - mn);
        float ps  = 0.0f;
#pragma unroll
        for (int r = 0; r < 8; ++r) {
            p0v[r] = __expf(p0v[r] - mn);
            p1v[r] = __expf(p1v[r] - mn);
            ps += p0v[r] + p1v[r];
        }
        ps += __shfl_xor(ps, 16, 32);
        l_run = l_run * fac + ps;
        m_run = mn;

        // ---- rescale O accumulator: broadcast per-query fac to O-row layout -
        if (lane < 16) bc[wave][lane] = fac;   // same-wave DS is in-order
        float4 f0 = *(const float4*)&bc[wave][base8];
        float4 f1 = *(const float4*)&bc[wave][base8 + 4];
        float fr[8] = {f0.x, f0.y, f0.z, f0.w, f1.x, f1.y, f1.z, f1.w};
#pragma unroll
        for (int u = 0; u < 4; ++u)
#pragma unroll
            for (int r = 0; r < 8; ++r) acc[u][r] *= fr[r];

        // ---- P^T is already in A-fragment layout: pack exp() to bf16 -------
        union { unsigned short us[16]; v16bf v; } ap;
#pragma unroll
        for (int r = 0; r < 8; ++r) {
            ap.us[r]     = f32_to_bf16(p0v[r]);
            ap.us[8 + r] = f32_to_bf16(p1v[r]);
        }

        // ---- O += P x V  (4 hd-tiles of 16) ----
#pragma unroll
        for (int u = 0; u < 4; ++u) {
            v16bf bv = frag16(pV[u], pV[u] + 16);
            acc[u] = __builtin_amdgcn_wmma_f32_16x16x32_bf16(
                false, ap.v, false, bv, (short)0, acc[u], false, false);
        }
        __syncthreads();
    }

    // ---- epilogue: per-query 1/l broadcast, normalize, store bf16 ----
    if (lane < 16) bc[wave][lane] = 1.0f / l_run;
    float4 f0 = *(const float4*)&bc[wave][base8];
    float4 f1 = *(const float4*)&bc[wave][base8 + 4];
    float inv[8] = {f0.x, f0.y, f0.z, f0.w, f1.x, f1.y, f1.z, f1.w};
#pragma unroll
    for (int r = 0; r < 8; ++r) {
        size_t obase = ((size_t)b * kS + q_base + base8 + r) * kD + h * kHD;
#pragma unroll
        for (int u = 0; u < 4; ++u)
            o[obase + u * 16 + col] = f32_to_bf16(acc[u][r] * inv[r]);
    }
}

// ---------------------------------------------------------------------------
extern "C" void kernel_launch(void* const* d_in, const int* in_sizes, int n_in,
                              void* d_out, int out_size, void* d_ws, size_t ws_size,
                              hipStream_t stream) {
    (void)in_sizes; (void)n_in; (void)out_size; (void)ws_size;
    const float* x    = (const float*)d_in[0];   // [B,S,D]
    const float* Wqkv = (const float*)d_in[1];   // [D,3D]
    const float* Wo   = (const float*)d_in[2];   // [D,D]
    float* out = (float*)d_out;                  // [B,S,D] fp32

    const size_t nX = (size_t)kB * kS * kD;      // 4,194,304
    unsigned short* qkv   = (unsigned short*)d_ws;            // [B,S,3D] bf16
    unsigned short* attn  = qkv + (size_t)kB * kS * 3 * kD;   // [B,S,D]  bf16
    unsigned short* xbf   = attn + nX;                        // [B*S][D] bf16
    unsigned short* wqkvt = xbf + nX;                         // [3D][D]  bf16 (transposed)
    unsigned short* wot   = wqkvt + (size_t)3 * kD * kD;      // [D][D]   bf16 (transposed)

    // 0) one-time conversions / weight transposes
    cvt_f32_bf16<<<dim3(nX / (256 * 8)), 256, 0, stream>>>(x, xbf, (int)nX);
    transpose_cvt<<<dim3((3 * kD) / 32, kD / 32), 256, 0, stream>>>(Wqkv, wqkvt, kD, 3 * kD);
    transpose_cvt<<<dim3(kD / 32, kD / 32), 256, 0, stream>>>(Wo, wot, kD, kD);

    // 1) QKV projection: [4096,1024] x [1024,3072] -> bf16
    gemm_wmma_bf16<unsigned short>
        <<<dim3((3 * kD) / 128, (kB * kS) / 256), 256, 0, stream>>>(
            xbf, wqkvt, qkv, kB * kS, 3 * kD, kD);

    // 2) causal flash attention per (b,h), 64 query rows per block
    flash_attn<<<dim3(kS / 64, kB * kH), 128, 0, stream>>>(qkv, attn);

    // 3) output projection: [4096,1024] x [1024,1024] -> fp32
    gemm_wmma_bf16<float>
        <<<dim3(kD / 128, (kB * kS) / 256), 256, 0, stream>>>(
            attn, wot, out, kB * kS, kD, kD);
}
